// DynamicHyperCell_42253888258467
// MI455X (gfx1250) — compile-verified, matrix operand order checked by reading, not run on hardware
//
#include <hip/hip_runtime.h>
#include <hip/hip_bf16.h>

// Dims from the reference
#define B_  64
#define T_  512
#define I_  64
#define H_  512
#define ZH_ 128
#define NZ_ 32
#define L_  3

typedef unsigned short u16;
typedef __attribute__((ext_vector_type(16))) __bf16 v16bf;
typedef __attribute__((ext_vector_type(8)))  float  v8f;

struct __attribute__((aligned(16))) U4 { unsigned x, y, z, w; };
union FragU { v16bf v; U4 q[2]; };

__device__ __forceinline__ u16 f2bf(float f) {
  unsigned u = __builtin_bit_cast(unsigned, f);
  u += 0x7fffu + ((u >> 16) & 1u);          // round-to-nearest-even
  return (u16)(u >> 16);
}
__device__ __forceinline__ float bf2f(u16 h) {
  unsigned u = ((unsigned)h) << 16;
  return __builtin_bit_cast(float, u);
}
__device__ __forceinline__ float sigm(float x) { return 1.0f / (1.0f + __expf(-x)); }

// Load one 16x32 bf16 fragment (A or B operand) given a pointer to this
// lane's first element: row*stride + k0 + half*8.  Elements 0..7 are
// K = khalf..khalf+7 (VGPRs 0..3), elements 8..15 are K+16 (VGPRs 4..7),
// matching the CDNA5 wave32 16-bit matrix layout.
__device__ __forceinline__ v16bf frag_ld(const u16* p) {
  FragU f;
  f.q[0] = *(const U4*)(p);
  f.q[1] = *(const U4*)(p + 16);
  return f.v;
}

#define WMMA_BF16(A, B, C) \
  __builtin_amdgcn_wmma_f32_16x16x32_bf16(false, (A), false, (B), (short)0, (C), false, false)

// ---------------- fp32 -> bf16 weight repack ----------------
__global__ void cvt_bf16_kernel(const float* __restrict__ s, u16* __restrict__ d, int n) {
  int i = blockIdx.x * blockDim.x + threadIdx.x;
  if (i < n) d[i] = f2bf(s[i]);
}

// ---------------- persistent hyper-GRU kernel ----------------
// One workgroup (32 wave32s) owns 16 batch rows and runs the whole T loop.
__global__ __launch_bounds__(1024, 1)
void hypergru_kernel(const float* __restrict__ xs,      // [B,T,I]
                     const float* __restrict__ cond,    // [B,I]
                     const float* __restrict__ h0,      // [1,B,H]
                     const float* __restrict__ hhat0,   // [1,B,ZH]
                     const float* __restrict__ x2h_b,   // [384]
                     const float* __restrict__ h2h_b,   // [384]
                     const float* __restrict__ zh_b,    // [96]
                     const float* __restrict__ zx_b,    // [96]
                     const float* __restrict__ db_b,    // [3,512]
                     const u16* __restrict__ w_x2h,     // [384,576] bf16
                     const u16* __restrict__ w_h2h,     // [384,128]
                     const u16* __restrict__ w_zh,      // [96,128]
                     const u16* __restrict__ w_zx,      // [96,128]
                     const u16* __restrict__ w_zb,      // [96,128]
                     const u16* __restrict__ w_dh,      // [3*512,32]
                     const u16* __restrict__ w_dx,      // [3*512,32]
                     const u16* __restrict__ w_db,      // [3*512,32]
                     const u16* __restrict__ w_wh,      // [3*512,512]
                     const u16* __restrict__ w_wx,      // [3*512,64]
                     float* __restrict__ out)           // [B*T*H | B*H | B*ZH]
{
  // bf16 activation buffers (A-matrices, row-major, 16B-aligned rows)
  __shared__ __align__(16) u16   sXhat[16 * 576];   // cols 0..511 = h(bf16), 512..575 = cond
  __shared__ __align__(16) float sHhat[16 * 128];   // fp32 carried hyper state
  __shared__ __align__(16) u16   sHhatB[16 * 128];  // bf16 mirror for WMMA
  __shared__ __align__(16) u16   sXt[16 * 64];      // x_t bf16
  __shared__ __align__(16) u16   sZ[3][16 * 96];    // zh/zx/zb bf16

  const int tid   = threadIdx.x;
  const int wave  = tid >> 5;
  const int lane  = tid & 31;
  const int r     = lane & 15;     // row (A/B) / N column (C/D)
  const int half  = lane >> 4;     // K-half for A/B, M-half for C/D
  const int bbase = blockIdx.x * 16;

  // ---------- init state from h_c / h_c_hat / cond ----------
  for (int i = tid; i < 16 * 512; i += 1024) {
    int m = i >> 9, c = i & 511;
    sXhat[m * 576 + c] = f2bf(h0[(size_t)(bbase + m) * H_ + c]);
  }
  for (int i = tid; i < 16 * 64; i += 1024) {
    int m = i >> 6, c = i & 63;
    sXhat[m * 576 + 512 + c] = f2bf(cond[(size_t)(bbase + m) * I_ + c]);
  }
  for (int i = tid; i < 16 * 128; i += 1024) {
    float v = hhat0[(size_t)bbase * ZH_ + i];
    sHhat[i]  = v;
    sHhatB[i] = f2bf(v);
  }

  // ---------- hoist per-lane loop invariants ----------
  // Phase A (waves 0..7): ZH tile jA, biases + weight row offsets
  const bool aAct = (wave < 8);
  const int  jA   = wave & 7;
  const int  ncA  = jA * 16 + r;
  const float brA  = x2h_b[ncA]       + h2h_b[ncA];
  const float bgA  = x2h_b[128 + ncA] + h2h_b[128 + ncA];
  const float binA = x2h_b[256 + ncA];
  const float bhnA = h2h_b[256 + ncA];
  const int offX2h = ncA * 576 + half * 8;   // + g*128*576, + k0
  const int offH2h = ncA * 128 + half * 8;   // + g*128*128, + k0

  // Phase B (waves 0..17)
  const int  pB  = wave / 6;
  const int  jB  = wave % 6;
  const int  ncB = jB * 16 + r;
  const u16* wzB = (pB == 0) ? w_zh : (pB == 1) ? w_zx : w_zb;
  const float biasB = (wave < 18) ? ((pB == 0) ? zh_b[ncB] : (pB == 1) ? zx_b[ncB] : 0.0f) : 0.0f;
  const int offZw = ncB * 128 + half * 8;

  // Phase C (all 32 waves): h-tile hs
  const int  hs  = wave * 16;
  const int  ncC = hs + r;
  const int offWh = ncC * 512 + half * 8;    // + l*512*512, + k0
  const int offWx = ncC * 64  + half * 8;    // + l*512*64,  + k0
  const int offD  = ncC * 32  + half * 8;    // + l*512*32
  const float bbC0 = db_b[0 * H_ + ncC];
  const float bbC1 = db_b[1 * H_ + ncC];
  const float bbC2 = db_b[2 * H_ + ncC];
  // LDS row bases
  const int rowXhat = r * 576 + half * 8;
  const int rowHhat = r * 128 + half * 8;
  const int rowXt   = r * 64  + half * 8;
  const int rowZ    = r * 96  + half * 8;    // + l*32

  __syncthreads();

  for (int t = 0; t < T_; ++t) {
    // ---- stage x_t (1024 threads == 16*64 elements) ----
    {
      int m = tid >> 6, c = tid & 63;
      sXt[m * 64 + c] = f2bf(xs[((size_t)(bbase + m) * T_ + t) * I_ + c]);
    }
    __syncthreads();

    // ================= Phase A: hyper GRU (waves 0..7) =================
    float hhNew[8];
    if (aAct) {
      v8f ai0 = {}, ai1 = {}, ai2 = {}, ah0 = {}, ah1 = {}, ah2 = {};
      for (int k0 = 0; k0 < 576; k0 += 32) {      // ih = x_hat @ x2h_w^T
        v16bf a  = frag_ld(&sXhat[rowXhat + k0]);
        v16bf b0 = frag_ld(&w_x2h[offX2h + 0 * 128 * 576 + k0]);
        v16bf b1 = frag_ld(&w_x2h[offX2h + 1 * 128 * 576 + k0]);
        v16bf b2 = frag_ld(&w_x2h[offX2h + 2 * 128 * 576 + k0]);
        ai0 = WMMA_BF16(a, b0, ai0);
        ai1 = WMMA_BF16(a, b1, ai1);
        ai2 = WMMA_BF16(a, b2, ai2);
      }
      for (int k0 = 0; k0 < 128; k0 += 32) {      // hh = hhat @ h2h_w^T
        v16bf a  = frag_ld(&sHhatB[rowHhat + k0]);
        v16bf b0 = frag_ld(&w_h2h[offH2h + 0 * 128 * 128 + k0]);
        v16bf b1 = frag_ld(&w_h2h[offH2h + 1 * 128 * 128 + k0]);
        v16bf b2 = frag_ld(&w_h2h[offH2h + 2 * 128 * 128 + k0]);
        ah0 = WMMA_BF16(a, b0, ah0);
        ah1 = WMMA_BF16(a, b1, ah1);
        ah2 = WMMA_BF16(a, b2, ah2);
      }
      for (int v = 0; v < 8; ++v) {
        int m = half * 8 + v;
        float rr = sigm(ai0[v] + ah0[v] + brA);
        float gg = sigm(ai1[v] + ah1[v] + bgA);
        float nn = tanhf(ai2[v] + binA + rr * (ah2[v] + bhnA));
        float old = sHhat[m * 128 + ncA];
        hhNew[v] = nn + gg * (old - nn);
      }
    }
    __syncthreads();                              // phase-A reads complete
    if (aAct) {
      for (int v = 0; v < 8; ++v) {
        int m = half * 8 + v;
        sHhat[m * 128 + ncA]  = hhNew[v];
        sHhatB[m * 128 + ncA] = f2bf(hhNew[v]);
        if (t == T_ - 1)
          out[(size_t)B_ * T_ * H_ + (size_t)B_ * H_ + (size_t)(bbase + m) * ZH_ + ncA] = hhNew[v];
      }
    }
    __syncthreads();

    // ============ Phase B: z projections (waves 0..17) ============
    if (wave < 18) {
      v8f acc = {};
      for (int k0 = 0; k0 < 128; k0 += 32) {
        v16bf a = frag_ld(&sHhatB[rowHhat + k0]);
        v16bf b = frag_ld(&wzB[offZw + k0]);
        acc = WMMA_BF16(a, b, acc);
      }
      for (int v = 0; v < 8; ++v) {
        int m = half * 8 + v;
        sZ[pB][m * 96 + ncB] = f2bf(acc[v] + biasB);
      }
    }
    __syncthreads();

    // ======== Phase C: main modulated GRU (all 32 waves, 1 h-tile each) ========
    // Gates share the same A matrices -> one A load feeds 3 accumulator chains.
    float r0[8], g0[8], t2a[8], t2b[8], hnew[8];
    {
      v8f hh0 = {}, hh1 = {}, hh2 = {};           // whh per gate
      for (int k0 = 0; k0 < 512; k0 += 32) {      // K over h (dominant loop)
        v16bf a  = frag_ld(&sXhat[rowXhat + k0]);
        v16bf b0 = frag_ld(&w_wh[offWh + 0 * 512 * 512 + k0]);
        v16bf b1 = frag_ld(&w_wh[offWh + 1 * 512 * 512 + k0]);
        v16bf b2 = frag_ld(&w_wh[offWh + 2 * 512 * 512 + k0]);
        hh0 = WMMA_BF16(a, b0, hh0);
        hh1 = WMMA_BF16(a, b1, hh1);
        hh2 = WMMA_BF16(a, b2, hh2);
      }
      v8f xx0 = {}, xx1 = {}, xx2 = {};           // wxx per gate
      for (int k0 = 0; k0 < 64; k0 += 32) {       // K over I
        v16bf a  = frag_ld(&sXt[rowXt + k0]);
        v16bf b0 = frag_ld(&w_wx[offWx + 0 * 512 * 64 + k0]);
        v16bf b1 = frag_ld(&w_wx[offWx + 1 * 512 * 64 + k0]);
        v16bf b2 = frag_ld(&w_wx[offWx + 2 * 512 * 64 + k0]);
        xx0 = WMMA_BF16(a, b0, xx0);
        xx1 = WMMA_BF16(a, b1, xx1);
        xx2 = WMMA_BF16(a, b2, xx2);
      }
      // dh/dx/db: K = NZ = 32, A = z slice for layer l
      v8f dh0 = {}, dh1 = {}, dh2 = {}, dx0 = {}, dx1 = {}, dx2 = {}, db0 = {}, db1 = {}, db2 = {};
      {
        v16bf ah0_ = frag_ld(&sZ[0][rowZ + 0 * 32]);
        v16bf ah1_ = frag_ld(&sZ[0][rowZ + 1 * 32]);
        v16bf ah2_ = frag_ld(&sZ[0][rowZ + 2 * 32]);
        v16bf bh0_ = frag_ld(&w_dh[offD + 0 * 512 * 32]);
        v16bf bh1_ = frag_ld(&w_dh[offD + 1 * 512 * 32]);
        v16bf bh2_ = frag_ld(&w_dh[offD + 2 * 512 * 32]);
        dh0 = WMMA_BF16(ah0_, bh0_, dh0);
        dh1 = WMMA_BF16(ah1_, bh1_, dh1);
        dh2 = WMMA_BF16(ah2_, bh2_, dh2);
        v16bf ax0_ = frag_ld(&sZ[1][rowZ + 0 * 32]);
        v16bf ax1_ = frag_ld(&sZ[1][rowZ + 1 * 32]);
        v16bf ax2_ = frag_ld(&sZ[1][rowZ + 2 * 32]);
        v16bf bx0_ = frag_ld(&w_dx[offD + 0 * 512 * 32]);
        v16bf bx1_ = frag_ld(&w_dx[offD + 1 * 512 * 32]);
        v16bf bx2_ = frag_ld(&w_dx[offD + 2 * 512 * 32]);
        dx0 = WMMA_BF16(ax0_, bx0_, dx0);
        dx1 = WMMA_BF16(ax1_, bx1_, dx1);
        dx2 = WMMA_BF16(ax2_, bx2_, dx2);
        v16bf ab0_ = frag_ld(&sZ[2][rowZ + 0 * 32]);
        v16bf ab1_ = frag_ld(&sZ[2][rowZ + 1 * 32]);
        v16bf ab2_ = frag_ld(&sZ[2][rowZ + 2 * 32]);
        v16bf bb0_ = frag_ld(&w_db[offD + 0 * 512 * 32]);
        v16bf bb1_ = frag_ld(&w_db[offD + 1 * 512 * 32]);
        v16bf bb2_ = frag_ld(&w_db[offD + 2 * 512 * 32]);
        db0 = WMMA_BF16(ab0_, bb0_, db0);
        db1 = WMMA_BF16(ab1_, bb1_, db1);
        db2 = WMMA_BF16(ab2_, bb2_, db2);
      }
      for (int v = 0; v < 8; ++v) {
        r0[v]  = sigm(dh0[v] * hh0[v] + dx0[v] * xx0[v] + db0[v] + bbC0);
        g0[v]  = sigm(dh1[v] * hh1[v] + dx1[v] * xx1[v] + db1[v] + bbC1);
        t2a[v] = dh2[v] * hh2[v];
        t2b[v] = dx2[v] * xx2[v] + db2[v] + bbC2;
      }
      for (int v = 0; v < 8; ++v) {
        int m = half * 8 + v;
        float n0   = tanhf(r0[v] * t2a[v] + t2b[v]);
        float hold = bf2f(sXhat[m * 576 + ncC]);
        hnew[v] = n0 + g0[v] * (hold - n0);
      }
    }
    __syncthreads();                              // phase-C reads complete
    {
      for (int v = 0; v < 8; ++v) {
        int m = half * 8 + v;
        sXhat[m * 576 + ncC] = f2bf(hnew[v]);
        out[((size_t)(bbase + m) * T_ + t) * H_ + ncC] = hnew[v];
        if (t == T_ - 1)
          out[(size_t)B_ * T_ * H_ + (size_t)(bbase + m) * H_ + ncC] = hnew[v];
      }
    }
    __syncthreads();
  }
}

extern "C" void kernel_launch(void* const* d_in, const int* in_sizes, int n_in,
                              void* d_out, int out_size, void* d_ws, size_t ws_size,
                              hipStream_t stream) {
  (void)in_sizes; (void)n_in; (void)out_size; (void)ws_size;
  const float* xs      = (const float*)d_in[0];
  const float* cond    = (const float*)d_in[1];
  const float* h_c     = (const float*)d_in[2];
  const float* h_c_hat = (const float*)d_in[3];
  const float* x2h_w   = (const float*)d_in[4];
  const float* x2h_b   = (const float*)d_in[5];
  const float* h2h_w   = (const float*)d_in[6];
  const float* h2h_b   = (const float*)d_in[7];
  const float* zh_w    = (const float*)d_in[8];
  const float* zh_b    = (const float*)d_in[9];
  const float* zx_w    = (const float*)d_in[10];
  const float* zx_b    = (const float*)d_in[11];
  const float* zb_w    = (const float*)d_in[12];
  const float* dh_w    = (const float*)d_in[13];
  const float* dx_w    = (const float*)d_in[14];
  const float* db_w    = (const float*)d_in[15];
  const float* db_b    = (const float*)d_in[16];
  const float* wh      = (const float*)d_in[17];
  const float* wx      = (const float*)d_in[18];
  float* out = (float*)d_out;

  // bf16 weight pool in d_ws (repacked every call; deterministic)
  u16* ws = (u16*)d_ws;
  size_t off = 0;
  u16* w_x2h = ws + off; off += (size_t)384 * 576;
  u16* w_h2h = ws + off; off += (size_t)384 * 128;
  u16* w_zh  = ws + off; off += (size_t)96 * 128;
  u16* w_zx  = ws + off; off += (size_t)96 * 128;
  u16* w_zb  = ws + off; off += (size_t)96 * 128;
  u16* w_dh  = ws + off; off += (size_t)3 * 512 * 32;
  u16* w_dx  = ws + off; off += (size_t)3 * 512 * 32;
  u16* w_db  = ws + off; off += (size_t)3 * 512 * 32;
  u16* w_wh  = ws + off; off += (size_t)3 * 512 * 512;
  u16* w_wx  = ws + off; off += (size_t)3 * 512 * 64;

  auto cvt = [&](const float* s, u16* d, int n) {
    cvt_bf16_kernel<<<(n + 255) / 256, 256, 0, stream>>>(s, d, n);
  };
  cvt(x2h_w, w_x2h, 384 * 576);
  cvt(h2h_w, w_h2h, 384 * 128);
  cvt(zh_w,  w_zh,  96 * 128);
  cvt(zx_w,  w_zx,  96 * 128);
  cvt(zb_w,  w_zb,  96 * 128);
  cvt(dh_w,  w_dh,  3 * 512 * 32);
  cvt(dx_w,  w_dx,  3 * 512 * 32);
  cvt(db_w,  w_db,  3 * 512 * 32);
  cvt(wh,    w_wh,  3 * 512 * 512);
  cvt(wx,    w_wx,  3 * 512 * 64);

  // 4 persistent workgroups (16 batch rows each), 32 wave32s per workgroup.
  hypergru_kernel<<<B_ / 16, 1024, 0, stream>>>(
      xs, cond, h_c, h_c_hat, x2h_b, h2h_b, zh_b, zx_b, db_b,
      w_x2h, w_h2h, w_zh, w_zx, w_zb, w_dh, w_dx, w_db, w_wh, w_wx, out);
}
